// Memorizer_57320633532846
// MI455X (gfx1250) — 8-bit matrix operand order repaired and checked by reading, compiled, not run on hardware
//
#include <hip/hip_runtime.h>

typedef __attribute__((ext_vector_type(8))) int v8i;

#define N_QUERIES 4096
#define N_MEM     32768
#define D_FEAT    8

constexpr int QT = 4;
constexpr int KT = 16;

__global__ void memorizer_init_kernel(unsigned* __restrict__ lastIdx) {
    int i = blockIdx.x * blockDim.x + threadIdx.x;
    if (i < N_QUERIES) lastIdx[i] = 0u;
}

__device__ __forceinline__ v8i onehot_pack(float4 f) {
    v8i r = {};
    r[0] = (int)(1u << (8u * (unsigned)f.x));
    r[1] = (int)(1u << (8u * (unsigned)f.y));
    r[2] = (int)(1u << (8u * (unsigned)f.z));
    r[3] = (int)(1u << (8u * (unsigned)f.w));
    return r;
}

__global__ __launch_bounds__(256) void memorizer_match_kernel(
        const float* __restrict__ x,
        const float* __restrict__ keys,
        unsigned* __restrict__ lastIdx) {
    const int wave = blockIdx.x * (blockDim.x >> 5) + (threadIdx.x >> 5);
    const int lane = threadIdx.x & 31;

    constexpr int nKeyChunks = N_MEM / (16 * KT);
    const int qGroup = wave / nKeyChunks;
    const int kChunk = wave % nKeyChunks;
    const int qBase  = qGroup * 16 * QT;
    const int kBase  = kChunk * 16 * KT;

    const bool hi = (lane >= 16);
    const int  r  = lane & 15;

    const int aoff = hi ? 4 : 0;
    v8i A[QT];
    #pragma unroll
    for (int t = 0; t < QT; ++t) {
        const float* qrow = x + (size_t)(qBase + t * 16 + r) * D_FEAT;
        float4 f;
        f.x = qrow[aoff + 0];
        f.y = qrow[aoff + 1];
        f.z = qrow[aoff + 2];
        f.w = qrow[aoff + 3];
        A[t] = onehot_pack(f);
    }

    const int koff = hi ? 4 : 0;
    const float4* kbasep =
        (const float4*)(keys + (size_t)(kBase + r) * D_FEAT + koff);
    float4 kf = kbasep[0];
    for (int kt = 0; kt < KT; ++kt) {
        const int kTile = kBase + kt * 16;
        v8i B = onehot_pack(kf);

        const int ktn = (kt + 1 < KT) ? (kt + 1) : kt;
        float4 kfn = kbasep[(size_t)ktn * 32];

        v8i D[QT];
        #pragma unroll
        for (int t = 0; t < QT; ++t) {
            v8i c = {};
            D[t] = __builtin_amdgcn_wmma_i32_16x16x64_iu8(
                       false, A[t], false, B, c, false, false);
        }

        #pragma unroll
        for (int t = 0; t < QT; ++t) {
            unsigned m = 0u;
            #pragma unroll
            for (int rr = 0; rr < 8; ++rr)
                m |= (D[t][rr] == D_FEAT) ? (1u << rr) : 0u;
            if (__ballot(m != 0u)) {
                const unsigned ki1 = (unsigned)(kTile + r) + 1u;
                #pragma unroll
                for (int rr = 0; rr < 8; ++rr) {
                    if (m & (1u << rr)) {
                        const int qi = qBase + t * 16 + rr + (hi ? 8 : 0);
                        atomicMax(&lastIdx[qi], ki1);
                    }
                }
            }
        }
        kf = kfn;
    }
}

__global__ void memorizer_finalize_kernel(
        const float* __restrict__ x,
        const float* __restrict__ mem_values,
        const float* __restrict__ w,
        const float* __restrict__ b,
        const unsigned* __restrict__ lastIdx,
        float* __restrict__ out) {
    int q = blockIdx.x * blockDim.x + threadIdx.x;
    if (q >= N_QUERIES) return;
    unsigned idx = lastIdx[q];
    float v;
    if (idx > 0u) {
        v = mem_values[idx - 1u];
    } else {
        const float* qrow = x + (size_t)q * D_FEAT;
        float acc = b[0];
        #pragma unroll
        for (int dd = 0; dd < D_FEAT; ++dd) acc += qrow[dd] * w[dd];
        v = acc;
    }
    out[q] = v;
}

extern "C" void kernel_launch(void* const* d_in, const int* in_sizes, int n_in,
                              void* d_out, int out_size, void* d_ws, size_t ws_size,
                              hipStream_t stream) {
    const float* x          = (const float*)d_in[0];
    const float* mem_keys   = (const float*)d_in[1];
    const float* mem_values = (const float*)d_in[2];
    const float* w          = (const float*)d_in[3];
    const float* b          = (const float*)d_in[4];
    float* out              = (float*)d_out;

    unsigned* lastIdx = (unsigned*)d_ws;

    memorizer_init_kernel<<<(N_QUERIES + 255) / 256, 256, 0, stream>>>(lastIdx);

    constexpr int wavesTotal =
        (N_QUERIES / (16 * QT)) * (N_MEM / (16 * KT));
    memorizer_match_kernel<<<wavesTotal / 8, 256, 0, stream>>>(x, mem_keys, lastIdx);

    memorizer_finalize_kernel<<<(N_QUERIES + 255) / 256, 256, 0, stream>>>(
        x, mem_values, w, b, lastIdx, out);
}
